// Hgru1_real_1d_59373627900322
// MI455X (gfx1250) — matrix operand order repaired and checked by reading, not compiled, not run on hardware
//
#include <hip/hip_runtime.h>
#include <hip/hip_bf16.h>

#define SEQ    2048
#define BATCH  8
#define EMBED  1024
#define INNER  2048
#define ROWS   (SEQ * BATCH)          // 16384 rows of the (seq,batch,·) matrix
#define CHANS  (BATCH * INNER)        // 16384 independent scan channels
#define CH     16                     // scan chunks
#define CLEN   (SEQ / CH)             // 128 steps per chunk
#define PHT    16                     // B tiles staged to LDS per phase
#define LN_EPS 1e-5f

typedef _Float16 v16h __attribute__((ext_vector_type(16)));
typedef _Float16 h8   __attribute__((ext_vector_type(8)));
typedef _Float16 h4   __attribute__((ext_vector_type(4)));
typedef float    v8f  __attribute__((ext_vector_type(8)));
typedef unsigned int u32x4 __attribute__((ext_vector_type(4)));
typedef int          i32x4 __attribute__((ext_vector_type(4)));
typedef int          i32x8 __attribute__((ext_vector_type(8)));

#if __has_builtin(__builtin_amdgcn_tensor_load_to_lds)
#define HAVE_TDM 1
#else
#define HAVE_TDM 0
#endif

#define WMMA(A, B, C) \
  __builtin_amdgcn_wmma_f32_16x16x32_f16(false, (A), false, (B), (short)0, (C), false, false)

__device__ inline unsigned int lds_off(const void* p) {
  return (unsigned int)(uintptr_t)p;   // LDS aperture: addr[31:0] is the LDS offset
}

#if HAVE_TDM
// TDM copy of n_tiles*512 contiguous f16 (one packed B-tile stream) into LDS.
// D# per ISA 8.3/8.4: count=1, type=2, data_size=1 (2B), 2-D tile 512 x n_tiles,
// dim0 stride 512 (fully contiguous), no padding/iterate/multicast.
__device__ inline void tdm_stage(const _Float16* src, unsigned int lds_byte,
                                 unsigned int n_tiles) {
  const unsigned long long ga = (unsigned long long)(uintptr_t)src;
  u32x4 g0;
  g0[0] = 1u;                                            // count=1, user mode
  g0[1] = lds_byte;                                      // lds_addr
  g0[2] = (unsigned int)ga;                              // global_addr[31:0]
  g0[3] = (unsigned int)((ga >> 32) & 0x1FFFFFFu) | (2u << 30);  // addr[56:32]|type=2
  const unsigned int d0 = 512u;                          // dim0 in elements
  i32x8 g1;
  g1[0] = (int)(1u << 16);                               // mask=0, data_size=1 (2B)
  g1[1] = (int)((d0 & 0xFFFFu) << 16);                   // tensor_dim0[15:0]
  g1[2] = (int)(((d0 >> 16) & 0xFFFFu) | ((n_tiles & 0xFFFFu) << 16)); // dim0 hi|dim1 lo
  g1[3] = (int)(((n_tiles >> 16) & 0xFFFFu) | (d0 << 16));             // dim1 hi|tile_dim0
  g1[4] = (int)(n_tiles & 0xFFFFu);                      // tile_dim1 (tile_dim2=0)
  g1[5] = (int)d0;                                       // tensor_dim0_stride[31:0]
  g1[6] = 0; g1[7] = 0;
  i32x4 g2 = {}; i32x4 g3 = {};
#if __clang_major__ >= 23
  i32x8 g4 = {};
  __builtin_amdgcn_tensor_load_to_lds(g0, g1, g2, g3, g4, 0);
#else
  __builtin_amdgcn_tensor_load_to_lds(g0, g1, g2, g3, 0);
#endif
}
#endif

// ---------------------------------------------------------------------------
// A-fragment loader: 16-bit A 16x32 lane layout (ISA 7.12.2): two 16-B runs.
__device__ inline v16h load_a_frag(const _Float16* __restrict__ p, int ld,
                                   int m0, int k0, int lane) {
  const int m  = m0 + (lane & 15);
  const int kb = k0 + ((lane >> 4) << 3);
  const _Float16* rp = p + (size_t)m * ld + kb;
  h8 a0 = *(const h8*)(rp);
  h8 a1 = *(const h8*)(rp + 16);
  v16h a;
#pragma unroll
  for (int i = 0; i < 8; ++i) { a[i] = a0[i]; a[i + 8] = a1[i]; }
  return a;
}

// ---------------------------------------------------------------------------
__global__ __launch_bounds__(256) void pack_x_kernel(
    const float* __restrict__ x, _Float16* __restrict__ xh, int n4) {
  int i = blockIdx.x * blockDim.x + threadIdx.x;
  if (i >= n4) return;
  const float4 v = ((const float4*)x)[i];
  h4 o; o[0] = (_Float16)v.x; o[1] = (_Float16)v.y;
        o[2] = (_Float16)v.z; o[3] = (_Float16)v.w;
  ((h4*)xh)[i] = o;
}

// Repack W (K x N row-major, f32) into per-(nt,kt)-tile B-fragment order, f16.
__global__ __launch_bounds__(256) void pack_w_kernel(
    const float* __restrict__ W, _Float16* __restrict__ P, int K, int N) {
  int i = blockIdx.x * blockDim.x + threadIdx.x;
  if (i >= K * N) return;
  const int e    = i & 15;
  const int L    = (i >> 4) & 31;
  const int tile = i >> 9;
  const int KT   = K >> 5;
  const int kt   = tile % KT;
  const int nt   = tile / KT;
  const int k    = kt * 32 + ((L >> 4) << 3) + (e & 7) + ((e >> 3) << 4);
  const int n    = nt * 16 + (L & 15);
  P[i] = (_Float16)W[(size_t)k * N + n];
}

// ---------------------------------------------------------------------------
// Fused gate GEMM. Block = 8 waves, all sharing one nt; B tiles staged to LDS
// (TDM if available) in phases of PHT k-tiles. Each wave owns 2 M-tiles x 3
// weights = 6 accumulators -> each LDS B fragment feeds 2 WMMAs, each A
// fragment feeds 3 WMMAs. Nonlinearities fused in the epilogue.
__global__ __launch_bounds__(256) void gates_gemm_kernel(
    const _Float16* __restrict__ xh,
    const _Float16* __restrict__ wiP, const _Float16* __restrict__ wfP,
    const _Float16* __restrict__ woP,
    const float* __restrict__ bi, const float* __restrict__ bf,
    const float* __restrict__ bo, const int* __restrict__ lb_ptr,
    float* __restrict__ u, float* __restrict__ lam, float* __restrict__ gate) {
  __shared__ __align__(128) _Float16 sB[3][PHT * 512];   // 48 KB
  const int lane = threadIdx.x & 31;
  const int wave = threadIdx.x >> 5;
  const int nt   = blockIdx.x;                           // 0..127
  const int m0   = (blockIdx.y * 8 + wave) * 32;         // 2 M-tiles per wave
  const int KT   = EMBED / 32;                           // 32

  v8f ci0 = {}, ci1 = {}, cf0 = {}, cf1 = {}, co0 = {}, co1 = {};

  for (int ph = 0; ph < KT / PHT; ++ph) {
    const int kt0 = ph * PHT;
    const size_t soff = ((size_t)(nt * KT + kt0)) << 9;  // halves
    __syncthreads();                                      // phase WAR guard
#if HAVE_TDM
    if (wave == 0) {
      tdm_stage(wiP + soff, lds_off(&sB[0][0]), PHT);
      tdm_stage(wfP + soff, lds_off(&sB[1][0]), PHT);
      tdm_stage(woP + soff, lds_off(&sB[2][0]), PHT);
      __builtin_amdgcn_s_wait_tensorcnt(0);
    }
#else
    {
      const h8* gI = (const h8*)(wiP + soff);
      const h8* gF = (const h8*)(wfP + soff);
      const h8* gO = (const h8*)(woP + soff);
      h8* dI = (h8*)&sB[0][0]; h8* dF = (h8*)&sB[1][0]; h8* dO = (h8*)&sB[2][0];
      for (int c = threadIdx.x; c < PHT * 64; c += 256) {
        dI[c] = gI[c]; dF[c] = gF[c]; dO[c] = gO[c];
      }
    }
#endif
    __syncthreads();

#pragma unroll 2
    for (int ktl = 0; ktl < PHT; ++ktl) {
      const int k0 = (kt0 + ktl) * 32;
      if (ktl + 1 < PHT) {   // prefetch next A rows
        __builtin_prefetch(xh + (size_t)(m0 + (lane & 15)) * EMBED + k0 + 32, 0, 1);
        __builtin_prefetch(xh + (size_t)(m0 + 16 + (lane & 15)) * EMBED + k0 + 32, 0, 1);
      }
      v16h a0 = load_a_frag(xh, EMBED, m0,      k0, lane);
      v16h a1 = load_a_frag(xh, EMBED, m0 + 16, k0, lane);
      v16h b  = *(const v16h*)(&sB[0][ktl << 9] + (lane << 4));
      ci0 = WMMA(a0, b, ci0); ci1 = WMMA(a1, b, ci1);
      b = *(const v16h*)(&sB[1][ktl << 9] + (lane << 4));
      cf0 = WMMA(a0, b, cf0); cf1 = WMMA(a1, b, cf1);
      b = *(const v16h*)(&sB[2][ktl << 9] + (lane << 4));
      co0 = WMMA(a0, b, co0); co1 = WMMA(a1, b, co1);
    }
  }

  const float lb  = (float)lb_ptr[0];
  const int   n   = (nt << 4) + (lane & 15);
  const float vbi = bi[n], vbf = bf[n], vbo = bo[n];
  const int mbase = m0 + ((lane >> 4) << 3);
#pragma unroll
  for (int t = 0; t < 2; ++t) {
    const v8f& vi = t ? ci1 : ci0;
    const v8f& vf = t ? cf1 : cf0;
    const v8f& vo = t ? co1 : co0;
#pragma unroll
    for (int r = 0; r < 8; ++r) {
      const size_t idx = (size_t)(mbase + t * 16 + r) * INNER + n;
      const float xi = vi[r] + vbi, xf = vf[r] + vbf, xo = vo[r] + vbo;
      const float inp = xi / (1.f + __expf(-xi));               // silu
      const float l   = lb + (1.f - lb) / (1.f + __expf(-xf));  // lambda
      u[idx]    = (1.f - l) * inp;
      lam[idx]  = l;
      gate[idx] = 1.f / (1.f + __expf(-xo));                    // out gate
    }
  }
}

// ---------------------------------------------------------------------------
// Chunked scan: phase1 per-chunk (prodLam, h) in parallel; phase2 16-step
// cross-chunk recurrence per channel; phase3 replay chunk from its carry.
__global__ __launch_bounds__(256) void scan_p1_kernel(
    const float* __restrict__ u, const float* __restrict__ lam,
    float* __restrict__ P, float* __restrict__ H) {
  const int g    = blockIdx.x * blockDim.x + threadIdx.x;  // 0..CHANS*CH-1
  const int chan = g & (CHANS - 1);
  const int c    = g >> 14;
  size_t idx = (size_t)c * CLEN * CHANS + chan;
  float h = 0.f, p = 1.f;
#pragma unroll 8
  for (int s = 0; s < CLEN; ++s) {
    const float l = lam[idx];
    h = fmaf(l, h, u[idx]);
    p *= l;
    idx += CHANS;
  }
  P[g] = p; H[g] = h;
}

__global__ __launch_bounds__(256) void scan_p2_kernel(
    const float* __restrict__ P, const float* __restrict__ H,
    float* __restrict__ S) {
  const int chan = blockIdx.x * blockDim.x + threadIdx.x;  // 0..CHANS-1
  float s = 0.f;
#pragma unroll
  for (int c = 0; c < CH; ++c) {
    S[c * CHANS + chan] = s;                               // carry INTO chunk c
    s = fmaf(P[c * CHANS + chan], s, H[c * CHANS + chan]);
  }
}

__global__ __launch_bounds__(256) void scan_p3_kernel(
    float* __restrict__ u, const float* __restrict__ lam,
    const float* __restrict__ S) {
  const int g    = blockIdx.x * blockDim.x + threadIdx.x;
  const int chan = g & (CHANS - 1);
  const int c    = g >> 14;
  size_t idx = (size_t)c * CLEN * CHANS + chan;
  float h = S[g];
#pragma unroll 8
  for (int s = 0; s < CLEN; ++s) {
    h = fmaf(lam[idx], h, u[idx]);
    u[idx] = h;
    idx += CHANS;
  }
}

// ---------------------------------------------------------------------------
// g = h*gate; LayerNorm over 2048 channels; emit f16 for the proj GEMM.
__global__ __launch_bounds__(256) void ln_kernel(
    const float* __restrict__ h, const float* __restrict__ gate,
    const float* __restrict__ gamma, const float* __restrict__ beta,
    _Float16* __restrict__ normed) {
  const int row = blockIdx.x;
  const size_t base = (size_t)row * INNER;
  const int t = threadIdx.x;

  float g[8], s1 = 0.f, s2 = 0.f;
#pragma unroll
  for (int i = 0; i < 8; ++i) {
    const int c = t + i * 256;
    const float v = h[base + c] * gate[base + c];
    g[i] = v; s1 += v; s2 += v * v;
  }
#pragma unroll
  for (int off = 16; off > 0; off >>= 1) {
    s1 += __shfl_down(s1, off, 32);
    s2 += __shfl_down(s2, off, 32);
  }
  __shared__ float red1[8], red2[8];
  if ((t & 31) == 0) { red1[t >> 5] = s1; red2[t >> 5] = s2; }
  __syncthreads();
  float tot1 = 0.f, tot2 = 0.f;
#pragma unroll
  for (int w = 0; w < 8; ++w) { tot1 += red1[w]; tot2 += red2[w]; }
  const float mu  = tot1 * (1.f / INNER);
  const float var = tot2 * (1.f / INNER) - mu * mu;
  const float inv = rsqrtf(var + LN_EPS);
#pragma unroll
  for (int i = 0; i < 8; ++i) {
    const int c = t + i * 256;
    normed[base + c] = (_Float16)((g[i] - mu) * inv * gamma[c] + beta[c]);
  }
}

// ---------------------------------------------------------------------------
// Output projection: 8 waves/block share nt; B staged to LDS in phases of PHT
// tiles; each wave owns 4 M-tiles -> each B fragment feeds 4 WMMAs.
__global__ __launch_bounds__(256) void proj_gemm_kernel(
    const _Float16* __restrict__ normed, const _Float16* __restrict__ wpP,
    const float* __restrict__ bp, float* __restrict__ out) {
  __shared__ __align__(128) _Float16 sB[PHT * 512];      // 16 KB
  const int lane = threadIdx.x & 31;
  const int wave = threadIdx.x >> 5;
  const int nt   = blockIdx.x;                           // 0..63
  const int m0   = (blockIdx.y * 8 + wave) * 64;         // 4 M-tiles per wave
  const int KT   = INNER / 32;                           // 64

  v8f a0c = {}, a1c = {}, a2c = {}, a3c = {};
  for (int ph = 0; ph < KT / PHT; ++ph) {
    const int kt0 = ph * PHT;
    const size_t soff = ((size_t)(nt * KT + kt0)) << 9;
    __syncthreads();
#if HAVE_TDM
    if (wave == 0) {
      tdm_stage(wpP + soff, lds_off(&sB[0]), PHT);
      __builtin_amdgcn_s_wait_tensorcnt(0);
    }
#else
    {
      const h8* gp = (const h8*)(wpP + soff);
      h8* dp = (h8*)&sB[0];
      for (int c = threadIdx.x; c < PHT * 64; c += 256) dp[c] = gp[c];
    }
#endif
    __syncthreads();

#pragma unroll 2
    for (int ktl = 0; ktl < PHT; ++ktl) {
      const int k0 = (kt0 + ktl) * 32;
      v16h b  = *(const v16h*)(&sB[ktl << 9] + (lane << 4));
      v16h a0 = load_a_frag(normed, INNER, m0,      k0, lane);
      v16h a1 = load_a_frag(normed, INNER, m0 + 16, k0, lane);
      v16h a2 = load_a_frag(normed, INNER, m0 + 32, k0, lane);
      v16h a3 = load_a_frag(normed, INNER, m0 + 48, k0, lane);
      a0c = WMMA(a0, b, a0c); a1c = WMMA(a1, b, a1c);
      a2c = WMMA(a2, b, a2c); a3c = WMMA(a3, b, a3c);
    }
  }

  const int   n   = (nt << 4) + (lane & 15);
  const float vb  = bp[n];
  const int mbase = m0 + ((lane >> 4) << 3);
#pragma unroll
  for (int t = 0; t < 4; ++t) {
    const v8f& acc = t == 0 ? a0c : t == 1 ? a1c : t == 2 ? a2c : a3c;
#pragma unroll
    for (int r = 0; r < 8; ++r)
      out[(size_t)(mbase + t * 16 + r) * EMBED + n] = acc[r] + vb;
  }
}

// ---------------------------------------------------------------------------
extern "C" void kernel_launch(void* const* d_in, const int* in_sizes, int n_in,
                              void* d_out, int out_size, void* d_ws, size_t ws_size,
                              hipStream_t stream) {
  const float* x     = (const float*)d_in[0];
  const float* Wi    = (const float*)d_in[1];
  const float* bi    = (const float*)d_in[2];
  const float* Wf    = (const float*)d_in[3];
  const float* bf    = (const float*)d_in[4];
  const float* Wo    = (const float*)d_in[5];
  const float* bo    = (const float*)d_in[6];
  const float* Wp    = (const float*)d_in[7];
  const float* bp    = (const float*)d_in[8];
  const float* gamma = (const float*)d_in[9];
  const float* beta  = (const float*)d_in[10];
  const int*   lb    = (const int*)d_in[11];
  float*       out   = (float*)d_out;

  char* ws = (char*)d_ws;
  _Float16* xh   = (_Float16*)ws;  ws += (size_t)ROWS * EMBED * 2;   // 32 MB
  _Float16* wiP  = (_Float16*)ws;  ws += (size_t)EMBED * INNER * 2;  //  4 MB
  _Float16* wfP  = (_Float16*)ws;  ws += (size_t)EMBED * INNER * 2;  //  4 MB
  _Float16* woP  = (_Float16*)ws;  ws += (size_t)EMBED * INNER * 2;  //  4 MB
  _Float16* wpP  = (_Float16*)ws;  ws += (size_t)INNER * EMBED * 2;  //  4 MB
  float*    u    = (float*)ws;     ws += (size_t)ROWS * INNER * 4;   // 128 MB (-> h)
  float*    lam  = (float*)ws;     ws += (size_t)ROWS * INNER * 4;   // 128 MB
  float*    gate = (float*)ws;     ws += (size_t)ROWS * INNER * 4;   // 128 MB
  _Float16* nrm  = (_Float16*)ws;  ws += (size_t)ROWS * INNER * 2;   //  64 MB
  float*    cP   = (float*)ws;     ws += (size_t)CHANS * CH * 4;     //   1 MB
  float*    cH   = (float*)ws;     ws += (size_t)CHANS * CH * 4;     //   1 MB
  float*    cS   = (float*)ws;     ws += (size_t)CHANS * CH * 4;     //   1 MB

  // 1) f16 conversion / weight repack into B-fragment layout.
  pack_x_kernel<<<ROWS * EMBED / 4 / 256, 256, 0, stream>>>(x, xh, ROWS * EMBED / 4);
  pack_w_kernel<<<EMBED * INNER / 256, 256, 0, stream>>>(Wi, wiP, EMBED, INNER);
  pack_w_kernel<<<EMBED * INNER / 256, 256, 0, stream>>>(Wf, wfP, EMBED, INNER);
  pack_w_kernel<<<EMBED * INNER / 256, 256, 0, stream>>>(Wo, woP, EMBED, INNER);
  pack_w_kernel<<<INNER * EMBED / 256, 256, 0, stream>>>(Wp, wpP, INNER, EMBED);

  // 2) fused gate GEMMs (WMMA, LDS/TDM-staged B) + nonlinear epilogue.
  gates_gemm_kernel<<<dim3(INNER / 16, ROWS / 32 / 8), 256, 0, stream>>>(
      xh, wiP, wfP, woP, bi, bf, bo, lb, u, lam, gate);

  // 3) chunk-parallel linear recurrence (in place: u -> h).
  scan_p1_kernel<<<CHANS * CH / 256, 256, 0, stream>>>(u, lam, cP, cH);
  scan_p2_kernel<<<CHANS / 256, 256, 0, stream>>>(cP, cH, cS);
  scan_p3_kernel<<<CHANS * CH / 256, 256, 0, stream>>>(u, lam, cS);

  // 4) gated output + LayerNorm -> f16 activations.
  ln_kernel<<<ROWS, 256, 0, stream>>>(u, gate, gamma, beta, nrm);

  // 5) output projection (WMMA, LDS/TDM-staged B) + bias.
  proj_gemm_kernel<<<dim3(EMBED / 16, ROWS / 64 / 8), 256, 0, stream>>>(
      nrm, wpP, bp, out);
}
